// NeuralBloch_21191368638761
// MI455X (gfx1250) — matrix operand order repaired and checked by reading, not compiled, not run on hardware
//
#include <hip/hip_runtime.h>
#include <hip/hip_bf16.h>

typedef _Float16 v8h  __attribute__((ext_vector_type(8)));
typedef _Float16 v16h __attribute__((ext_vector_type(16)));
typedef float    v8f  __attribute__((ext_vector_type(8)));

#define Bn   2048
#define Tn   2048
#define HIDN 128
#define DIMY 3
#define DIMU 4
#define DIMP 5
#define ROWS 16      // batch rows per block (one WMMA M-tile)
#define NTHREADS 256 // 8 wave32 waves

union V16U { v16h v; v8h h[2]; };

// A-matrix (16x32 f16) pack per ISA 7.12.2:
// lane<16: halves 0-7 = K0..7,  halves 8-15 = K16..23
// lane>=16: halves 0-7 = K8..15, halves 8-15 = K24..31
__device__ __forceinline__ v16h pack_a(const _Float16* buf, int rstride, int row, int k0, int lane) {
    const _Float16* pr = buf + row * rstride + k0 + ((lane & 16) ? 8 : 0);
    V16U u;
    u.h[0] = *(const v8h*)(pr);
    u.h[1] = *(const v8h*)(pr + 16);
    return u.v;
}

// B-matrix (32x16 f16) pack: n = lane&15; lane<16 holds K=0..7,16..23, lane>=16 holds K=8..15,24..31
// (buf stored transposed [n][k] so each lane reads two contiguous 8-half runs)
__device__ __forceinline__ v16h pack_b(const _Float16* buf, int rstride, int n, int k0, int lane) {
    const _Float16* pr = buf + n * rstride + k0 + ((lane & 16) ? 8 : 0);
    V16U u;
    u.h[0] = *(const v8h*)(pr);
    u.h[1] = *(const v8h*)(pr + 16);
    return u.v;
}

// Branch-free tanh: 1 - 2/(exp2(2*log2e*x)+1); saturates via exp2->inf/0.
__device__ __forceinline__ float fast_tanh(float x) {
    float e = __builtin_amdgcn_exp2f(x * 2.8853900817779268f); // 2*log2(e)
    return 1.0f - 2.0f * __builtin_amdgcn_rcpf(e + 1.0f);
}

__global__ __launch_bounds__(NTHREADS)
void NeuralBloch_rk4_wmma(const float* __restrict__ y0,
                          const float* __restrict__ u,
                          const float* __restrict__ p,
                          const float* __restrict__ W1, const float* __restrict__ b1,
                          const float* __restrict__ W2, const float* __restrict__ b2,
                          const float* __restrict__ W3, const float* __restrict__ b3,
                          float* __restrict__ out)
{
    __shared__ __align__(16) _Float16 sW1t[HIDN * 32];    // [n][k], k padded 13->32 with 0
    __shared__ __align__(16) _Float16 sW2t[HIDN * HIDN];  // [n][k]
    __shared__ __align__(16) _Float16 sW3t[16 * HIDN];    // [n][k], n padded 3->16 with 0
    __shared__ __align__(16) _Float16 sX  [ROWS * 32];    // MLP input, K padded to 32
    __shared__ __align__(16) _Float16 sHa [ROWS * HIDN];  // hidden layer 1
    __shared__ __align__(16) _Float16 sHb [ROWS * HIDN];  // hidden layer 2
    __shared__ float sU[2][2][ROWS][DIMU];                // double-buffered interval endpoints
    __shared__ float sB1[HIDN], sB2[HIDN], sB3[16];

    const int tid  = threadIdx.x;
    const int lane = tid & 31;
    const int wv   = tid >> 5;
    const int row0 = blockIdx.x * ROWS;
    const float dt = 1.0f / (float)(Tn - 1);

    // ---- one-time init: transpose weights to f16 [n][k] in LDS ----
    for (int idx = tid; idx < HIDN * 32; idx += NTHREADS) {
        int n = idx >> 5, kk = idx & 31;
        sW1t[idx] = (kk < DIMY + DIMU + DIMP + 1) ? (_Float16)W1[kk * HIDN + n] : (_Float16)0.0f;
    }
    for (int idx = tid; idx < HIDN * HIDN; idx += NTHREADS) {
        int n = idx >> 7, kk = idx & 127;
        sW2t[idx] = (_Float16)W2[kk * HIDN + n];
    }
    for (int idx = tid; idx < 16 * HIDN; idx += NTHREADS) {
        int n = idx >> 7, kk = idx & 127;
        sW3t[idx] = (n < DIMY) ? (_Float16)W3[kk * DIMY + n] : (_Float16)0.0f;
    }
    for (int idx = tid; idx < HIDN; idx += NTHREADS) { sB1[idx] = b1[idx]; sB2[idx] = b2[idx]; }
    if (tid < 16) sB3[tid] = (tid < DIMY) ? b3[tid] : 0.0f;

    // static columns of x: p (k=7..11), zero pad (k=13..31)
    for (int idx = tid; idx < ROWS * 32; idx += NTHREADS) {
        int m = idx >> 5, k = idx & 31;
        if (k >= 7 && k != 12) {
            float v = (k < 12) ? p[(size_t)(row0 + m) * DIMP + (k - 7)] : 0.0f;
            sX[idx] = (_Float16)v;
        }
    }
    // u endpoints for interval 0 into buffer 0
    if (tid < 2 * ROWS) {
        int which = tid >> 4, m = tid & 15;
        float4 uv = *(const float4*)(u + ((size_t)(row0 + m) * Tn + which) * DIMU);
        sU[0][which][m][0] = uv.x; sU[0][which][m][1] = uv.y;
        sU[0][which][m][2] = uv.z; sU[0][which][m][3] = uv.w;
    }

    // ---- wave-0 owned RK state in D-layout registers ----
    const int mlo = (lane & 16) ? 8 : 0;   // M offset for this lane half
    const int nn  = lane & 15;             // N (output column) for this lane
    float yreg[8], kacc[8];
    if (wv == 0) {
        #pragma unroll
        for (int r = 0; r < 8; ++r) {
            int m = r + mlo;
            float v = 0.0f;
            if (nn < DIMY) v = y0[(size_t)(row0 + m) * DIMY + nn];
            yreg[r] = v;
            kacc[r] = 0.0f;
            if (nn < DIMY) {
                sX[m * 32 + nn] = (_Float16)v;                       // y columns of x
                out[((size_t)(row0 + m) * Tn + 0) * DIMY + nn] = v;  // trajectory point 0
            }
        }
    }
    __syncthreads();

    // ---- time marching: one RK4 step per grid interval ----
    for (int i = 0; i < Tn - 1; ++i) {
        const int cur = i & 1;
        const float ti = (float)i * dt;

        for (int s = 0; s < 4; ++s) {
            const float wgt = (s == 0) ? 0.0f : (s == 3 ? 1.0f : 0.5f);
            const float tq  = ti + ((s == 0) ? 0.0f : (s == 3 ? dt : 0.5f * dt));

            // stage-top dynamic x columns (wave-uniform work; y cols written by wave0
            // at the previous stage's end, after B_h2)
            if (wv == 2 || wv == 3) {            // u(tq): 64 elements, waves 2-3
                int idx = (wv - 2) * 32 + lane;
                int m = idx >> 2, c = idx & 3;
                float a0 = sU[cur][0][m][c], a1 = sU[cur][1][m][c];
                sX[m * 32 + 3 + c] = (_Float16)((1.0f - wgt) * a0 + wgt * a1);
            } else if (wv == 1) {                // tq column
                if (lane < 16) sX[lane * 32 + 12] = (_Float16)tq;
            }
            __syncthreads();  // B_entry: x ready

            // prefetch next interval's u endpoints (overlaps with WMMA work below)
            if (s == 0 && wv == 4 && (i + 1) < Tn - 1) {
                int which = lane >> 4, m = lane & 15;
                float4 uv = *(const float4*)(u + ((size_t)(row0 + m) * Tn + (i + 1 + which)) * DIMU);
                sU[cur ^ 1][which][m][0] = uv.x; sU[cur ^ 1][which][m][1] = uv.y;
                sU[cur ^ 1][which][m][2] = uv.z; sU[cur ^ 1][which][m][3] = uv.w;
            }

            // ---- layer 1: (16x32) @ (32x128) -> wave wv owns cols [16wv,16wv+16) ----
            {
                const int ng = wv * 16 + nn;
                v16h a = pack_a(sX, 32, nn, 0, lane);
                v16h bm = pack_b(sW1t, 32, ng, 0, lane);
                v8f acc = {};
                acc = __builtin_amdgcn_wmma_f32_16x16x32_f16(false, a, false, bm, (short)0, acc, false, false);
                const float bias = sB1[ng];
                float th[8];
                #pragma unroll
                for (int r = 0; r < 8; ++r) th[r] = fast_tanh(acc[r] + bias);
                #pragma unroll
                for (int r = 0; r < 8; ++r) sHa[(r + mlo) * HIDN + ng] = (_Float16)th[r];
            }
            __syncthreads();  // B_h1

            // ---- layer 2: (16x128) @ (128x128), K split into 4 chunks of 32 ----
            {
                const int ng = wv * 16 + nn;
                v8f acc = {};
                #pragma unroll
                for (int kc = 0; kc < 4; ++kc) {
                    v16h a = pack_a(sHa, HIDN, nn, kc * 32, lane);
                    v16h bm = pack_b(sW2t, HIDN, ng, kc * 32, lane);
                    acc = __builtin_amdgcn_wmma_f32_16x16x32_f16(false, a, false, bm, (short)0, acc, false, false);
                }
                const float bias = sB2[ng];
                float th[8];
                #pragma unroll
                for (int r = 0; r < 8; ++r) th[r] = fast_tanh(acc[r] + bias);
                #pragma unroll
                for (int r = 0; r < 8; ++r) sHb[(r + mlo) * HIDN + ng] = (_Float16)th[r];
            }
            __syncthreads();  // B_h2

            // ---- layer 3 (16x128)@(128x3-pad16) + RK update: wave 0 only ----
            if (wv == 0) {
                v8f acc = {};
                #pragma unroll
                for (int kc = 0; kc < 4; ++kc) {
                    v16h a = pack_a(sHb, HIDN, nn, kc * 32, lane);
                    v16h bm = pack_b(sW3t, HIDN, nn, kc * 32, lane);
                    acc = __builtin_amdgcn_wmma_f32_16x16x32_f16(false, a, false, bm, (short)0, acc, false, false);
                }
                const float bias = sB3[nn];
                float dy[8];
                #pragma unroll
                for (int r = 0; r < 8; ++r) dy[r] = acc[r] + bias;

                if (s < 3) {
                    const float cs = (s == 0) ? 1.0f : 2.0f;    // k-accumulation weight
                    const float as = (s == 2) ? 1.0f : 0.5f;    // next-stage offset
                    #pragma unroll
                    for (int r = 0; r < 8; ++r) {
                        kacc[r] += cs * dy[r];
                        if (nn < DIMY)
                            sX[(r + mlo) * 32 + nn] = (_Float16)(yreg[r] + dt * as * dy[r]);
                    }
                } else {
                    #pragma unroll
                    for (int r = 0; r < 8; ++r) {
                        kacc[r] += dy[r];
                        yreg[r] += (dt / 6.0f) * kacc[r];
                        kacc[r] = 0.0f;
                        if (nn < DIMY) {
                            sX[(r + mlo) * 32 + nn] = (_Float16)yreg[r];
                            out[((size_t)(row0 + r + mlo) * Tn + (i + 1)) * DIMY + nn] = yreg[r];
                        }
                    }
                }
            }
        }
    }
}

extern "C" void kernel_launch(void* const* d_in, const int* in_sizes, int n_in,
                              void* d_out, int out_size, void* d_ws, size_t ws_size,
                              hipStream_t stream) {
    // setup_inputs order: y0, t, u, p, W1, b1, W2, b2, W3, b3  (all float32)
    const float* y0 = (const float*)d_in[0];
    // d_in[1] is t (linspace 0..1) -- recomputed on device from Tn
    const float* u  = (const float*)d_in[2];
    const float* p  = (const float*)d_in[3];
    const float* W1 = (const float*)d_in[4];
    const float* b1 = (const float*)d_in[5];
    const float* W2 = (const float*)d_in[6];
    const float* b2 = (const float*)d_in[7];
    const float* W3 = (const float*)d_in[8];
    const float* b3 = (const float*)d_in[9];
    float* out = (float*)d_out;

    dim3 grid(Bn / ROWS);   // 128 blocks, 16 trajectories each
    dim3 block(NTHREADS);   // 8 wave32 waves: hidden cols split across waves
    NeuralBloch_rk4_wmma<<<grid, block, 0, stream>>>(y0, u, p, W1, b1, W2, b2, W3, b3, out);
}
